// EfficientGlobalPointer_62740882260566
// MI455X (gfx1250) — matrix operand order repaired and checked by reading, not compiled, not run on hardware
//
#include <hip/hip_runtime.h>

typedef _Float16 v16h __attribute__((ext_vector_type(16)));
typedef _Float16 v8h  __attribute__((ext_vector_type(8)));
typedef _Float16 h2   __attribute__((ext_vector_type(2)));
typedef float    v8f  __attribute__((ext_vector_type(8)));
typedef float    v4f  __attribute__((ext_vector_type(4)));

#define HID   1024
#define SEQ   1024
#define BATCH 8
#define NHEAD 12
#define DHEAD 64

// ---------------------------------------------------------------------------
// K0: pre-swizzle W1 (1024x128 f32) into f16 WMMA B-fragment order.
// Bpre[((nt*32 + ks)*32 + lane)*16 + i] = f16(W1[(ks*32 + kb + i)*128 + nt*16 + ln])
//   lane<16 : kb=0 (K=0..15), lane>=16 : kb=16 (K=16..31); N = nt*16 + (lane&15)
// ---------------------------------------------------------------------------
__global__ void k0_preswizzle_w1(const float* __restrict__ W1,
                                 _Float16* __restrict__ Bpre) {
  int t = blockIdx.x * blockDim.x + threadIdx.x;   // 0..8191
  int nt   = t >> 10;          // 0..7  (N-tile)
  int rest = t & 1023;
  int ks   = rest >> 5;        // 0..31 (K-step of 32)
  int lane = rest & 31;
  int ln   = lane & 15;
  int kb   = (lane >> 4) << 4; // 0 or 16
  _Float16* dst = Bpre + (size_t)t * 16;
  const float* src = W1 + (size_t)(ks * 32 + kb) * 128 + nt * 16 + ln;
#pragma unroll
  for (int i = 0; i < 16; ++i) dst[i] = (_Float16)src[(size_t)i * 128];
}

// ---------------------------------------------------------------------------
// K1: H = inputs @ W1 + b1   (8192 x 1024 x 128), one wave per 16x16 tile.
// ---------------------------------------------------------------------------
__global__ void k1_h_gemm(const float* __restrict__ A,
                          const _Float16* __restrict__ Bpre,
                          const float* __restrict__ b1,
                          float* __restrict__ H) {
  int w    = blockIdx.x * (blockDim.x >> 5) + (threadIdx.x >> 5); // 0..4095
  int lane = threadIdx.x & 31;
  int mt = w >> 3, nt = w & 7;
  int m0 = mt * 16, n0 = nt * 16;
  int ln  = lane & 15;
  int hi  = lane >> 4;       // 0/1
  int kbA = hi << 4;         // A fragment K-base: 0 or 16

  const float* arow = A + (size_t)(m0 + ln) * HID;
  v8f acc = {};
  for (int ks = 0; ks < 32; ++ks) {
    int k0 = ks * 32;
    // A fragment: halves 0..15 = K k0+kbA+0..15
    v4f a0 = *(const v4f*)(arow + k0 + kbA);
    v4f a1 = *(const v4f*)(arow + k0 + kbA + 4);
    v4f a2 = *(const v4f*)(arow + k0 + kbA + 8);
    v4f a3 = *(const v4f*)(arow + k0 + kbA + 12);
    v16h af;
#pragma unroll
    for (int i = 0; i < 4; ++i) {
      af[i]      = (_Float16)a0[i];
      af[i + 4]  = (_Float16)a1[i];
      af[i + 8]  = (_Float16)a2[i];
      af[i + 12] = (_Float16)a3[i];
    }
    v16h bf = *(const v16h*)(Bpre + (((size_t)(nt * 32 + ks) * 32 + lane) << 4));
    acc = __builtin_amdgcn_wmma_f32_16x16x32_f16(false, af, false, bf,
                                                 (short)0, acc, false, false);
  }
  // C/D layout: VGPR r -> M = m0 + hi*8 + r, N = n0 + ln
  float bb = b1[n0 + ln];
  int mbase = m0 + hi * 8;
#pragma unroll
  for (int r = 0; r < 8; ++r)
    H[(size_t)(mbase + r) * 128 + n0 + ln] = acc[r] + bb;
}

// ---------------------------------------------------------------------------
// K2: per-row RoPE(q), RoPE(k) -> f16, and Bias = (H @ W2 + b2)/2.
// One wave per row (8192 rows). h row = [q0,k0,q1,k1, ...] interleaved.
// ---------------------------------------------------------------------------
__global__ void k2_rope_bias(const float* __restrict__ H,
                             const float* __restrict__ W2,
                             const float* __restrict__ b2,
                             _Float16* __restrict__ QR,
                             _Float16* __restrict__ KR,
                             float* __restrict__ Bias) {
  int w    = blockIdx.x * (blockDim.x >> 5) + (threadIdx.x >> 5); // row 0..8191
  int lane = threadIdx.x & 31;
  int pos  = w & (SEQ - 1);
  const float* hrow = H + (size_t)w * 128;

  // RoPE pair j (lane j handles q/k pair j, j = 0..31)
  {
    int j = lane;
    v4f hv = *(const v4f*)(hrow + 4 * j);   // {q[2j], k[2j], q[2j+1], k[2j+1]}
    float inv = __expf(-(float)(2 * j) * (9.210340371976184f / 64.0f));
    float ang = (float)pos * inv;
    float c = __cosf(ang), s = __sinf(ang);
    float q0 = hv[0], k0 = hv[1], q1 = hv[2], k1 = hv[3];
    h2 qo, ko;
    qo[0] = (_Float16)(q0 * c - q1 * s);
    qo[1] = (_Float16)(q1 * c + q0 * s);
    ko[0] = (_Float16)(k0 * c - k1 * s);
    ko[1] = (_Float16)(k1 * c + k0 * s);
    *(h2*)(QR + (size_t)w * DHEAD + 2 * j) = qo;
    *(h2*)(KR + (size_t)w * DHEAD + 2 * j) = ko;
  }
  // Bias: lanes 0..23 compute one output channel each
  if (lane < 24) {
    float acc = b2[lane];
    for (int c = 0; c < 128; ++c) acc += hrow[c] * W2[c * 24 + lane];
    Bias[(size_t)w * 24 + lane] = 0.5f * acc;
  }
}

// ---------------------------------------------------------------------------
// K3: TRANSPOSED att tile via WMMA: D[n][m] = att[m][n] = sum_d q[m,d]*k[n,d]
//     (A = k-tile, B = q-tile; both fragments are contiguous row-major loads).
// Each lane then holds 8 CONSECUTIVE n for a fixed m -> two b128 NT stores
// per head per lane (full cache lines) instead of 8 scalar stores.
// One wave per (b, mt, nt) tile: 8*64*64 = 32768 waves.
// ---------------------------------------------------------------------------
__global__ void k3_att_out(const _Float16* __restrict__ QR,
                           const _Float16* __restrict__ KR,
                           const float* __restrict__ Bias,
                           const float* __restrict__ AM,
                           float* __restrict__ out) {
  int w    = blockIdx.x * (blockDim.x >> 5) + (threadIdx.x >> 5);
  int lane = threadIdx.x & 31;
  int b   = w >> 12;
  int rem = w & 4095;
  int mt = rem >> 6, nt = rem & 63;
  int m0 = mt * 16, n0 = nt * 16;
  int ln  = lane & 15;
  int hi  = lane >> 4;
  int kbA = hi << 4;   // A K-base 0/16
  int kbB = hi << 4;   // B K-base 0/16

  // A row index = n (A = k-tile), B column index = m (B = q-tile^T layout)
  const _Float16* arow = KR + (size_t)(b * SEQ + n0 + ln) * DHEAD;
  const _Float16* brow = QR + (size_t)(b * SEQ + m0 + ln) * DHEAD;

  v8f acc = {};
#pragma unroll
  for (int kk = 0; kk < DHEAD; kk += 32) {
    v8h alo = *(const v8h*)(arow + kk + kbA);
    v8h ahi = *(const v8h*)(arow + kk + kbA + 8);
    v16h af;
#pragma unroll
    for (int i = 0; i < 8; ++i) { af[i] = alo[i]; af[i + 8] = ahi[i]; }
    v16h bf = *(const v16h*)(brow + kk + kbB);   // B[d][m] = QR[m][d]
    acc = __builtin_amdgcn_wmma_f32_16x16x32_f16(false, af, false, bf,
                                                 (short)0, acc, false, false);
  }

  // D layout: VGPR r -> n = n0 + hi*8 + r (consecutive!), lane -> m = m0 + ln
  const float scale = 0.125f;   // 1/sqrt(64)
  int m     = m0 + ln;
  int nbase = n0 + hi * 8;
  float amm = AM[b * SEQ + m];

  // base[r] = att*scale - attention-mask term (pre-bias, pre-edge, pre-tril)
  float base[8];
#pragma unroll
  for (int r = 0; r < 8; ++r) {
    float amn = AM[b * SEQ + nbase + r];
    base[r] = acc[r] * scale - (1.0f - amm * amn) * 1.0e12f;
  }

  const float* biasB = Bias + (size_t)b * SEQ * 24;
  const float* bqrow = biasB + (size_t)m * 24;   // lane-fixed row bias source

  for (int h = 0; h < NHEAD; ++h) {
    float bq = bqrow[2 * h];                     // bias[b, 2h, m] — 1 load/head
    float bk[8];
#pragma unroll
    for (int r = 0; r < 8; ++r)                  // bias[b, 2h+1, n]
      bk[r] = biasB[(size_t)(nbase + r) * 24 + 2 * h + 1];

    v4f olo, ohi;
#pragma unroll
    for (int r = 0; r < 8; ++r) {
      int n = nbase + r;
      float v = base[r] + bq + bk[r];
      if (m == 0 || n == 0) v = -10000.0f;       // row-0 / col-0 sets
      if (n < m) v -= 1.0e12f;                   // strict lower-tri mask
      if (r < 4) olo[r] = v; else ohi[r - 4] = v;
    }
    float* op = out + (((size_t)(b * NHEAD + h) * SEQ + m) * SEQ + nbase);
    __builtin_nontemporal_store(olo, (v4f*)op);        // 16B, 32B-aligned base
    __builtin_nontemporal_store(ohi, (v4f*)(op + 4));
  }
}

// ---------------------------------------------------------------------------
extern "C" void kernel_launch(void* const* d_in, const int* in_sizes, int n_in,
                              void* d_out, int out_size, void* d_ws, size_t ws_size,
                              hipStream_t stream) {
  const float* inputs = (const float*)d_in[0];   // (8,1024,1024)
  const float* am     = (const float*)d_in[1];   // (8,1024)
  const float* W1     = (const float*)d_in[2];   // (1024,128)
  const float* b1     = (const float*)d_in[3];   // (128,)
  const float* W2     = (const float*)d_in[4];   // (128,24)
  const float* b2     = (const float*)d_in[5];   // (24,)
  float* out = (float*)d_out;                    // (8,12,1024,1024)

  char* ws = (char*)d_ws;
  _Float16* Bpre = (_Float16*)(ws);                            // 256 KB
  float*    H    = (float*)(ws + (256 << 10));                 // 4 MB
  _Float16* QR   = (_Float16*)(ws + (256 << 10) + (4 << 20));  // 1 MB
  _Float16* KR   = (_Float16*)(ws + (256 << 10) + (5 << 20));  // 1 MB
  float*    Bias = (float*)(ws + (256 << 10) + (6 << 20));     // 768 KB

  k0_preswizzle_w1<<<32, 256, 0, stream>>>(W1, Bpre);
  k1_h_gemm       <<<1024, 128, 0, stream>>>(inputs, Bpre, b1, H);
  k2_rope_bias    <<<2048, 128, 0, stream>>>(H, W2, b2, QR, KR, Bias);
  k3_att_out      <<<8192, 128, 0, stream>>>(QR, KR, Bias, am, out);
}
